// MiniCPM3AttentionMLA_41042707480911
// MI455X (gfx1250) — matrix operand order repaired and checked by reading, not compiled, not run on hardware
//
#include <hip/hip_runtime.h>
#include <hip/hip_bf16.h>

typedef __attribute__((ext_vector_type(16))) __bf16 v16bf;
typedef __attribute__((ext_vector_type(8)))  float  v8f;

#define T_      2048
#define HID_    2560
#define H_      40
#define NOPE_   64
#define ROPE_   32
#define V_      64
#define QLORA_  768
#define KVLORA_ 256
#define DQ_     96      /* NOPE+ROPE */
#define DKV_    288     /* KVLORA+ROPE */
#define SCALE_  0.10206207261596577f   /* 96^-0.5 */

// ---------------------------------------------------------------------------
// float -> bf16 round-to-nearest-even (software)
__device__ __forceinline__ __bf16 tobf(float f) {
  unsigned u = __float_as_uint(f);
  unsigned r = (u + 0x7FFFu + ((u >> 16) & 1u)) >> 16;
  unsigned short hs = (unsigned short)r;
  __bf16 b;
  __builtin_memcpy(&b, &hs, sizeof(b));
  return b;
}

// Build a 16-element bf16 fragment from two 16-byte (8 x bf16) runs.
union FragU { uint4 q[2]; v16bf v; };
__device__ __forceinline__ v16bf ld_frag(const __bf16* p0, const __bf16* p1) {
  FragU f;
  f.q[0] = *reinterpret_cast<const uint4*>(p0);
  f.q[1] = *reinterpret_cast<const uint4*>(p1);
  return f.v;
}

// ---------------------------------------------------------------------------
// f32 -> bf16 elementwise (row-major preserved). n4 = elems/4.
__global__ void __launch_bounds__(256)
cvt_rm(const float* __restrict__ src, __bf16* __restrict__ dst, long n4) {
  long i = (long)blockIdx.x * 256 + threadIdx.x;
  if (i >= n4) return;
  float4 f = reinterpret_cast<const float4*>(src)[i];
  __bf16* d = dst + i * 4;
  d[0] = tobf(f.x); d[1] = tobf(f.y); d[2] = tobf(f.z); d[3] = tobf(f.w);
}

// f32 (R x C, row-major) -> bf16 (C x R) transposed, LDS-tiled, batched over z.
// grid = (C/32, R/32, batch), block = 256 (32x8).
__global__ void __launch_bounds__(256)
cvt_tr(const float* __restrict__ src, __bf16* __restrict__ dst,
       int R, int C, long sbat, long dbat) {
  __shared__ float tile[32][33];
  src += (long)blockIdx.z * sbat;
  dst += (long)blockIdx.z * dbat;
  const int c0 = blockIdx.x * 32, r0 = blockIdx.y * 32;
  const int tx = threadIdx.x & 31, ty = threadIdx.x >> 5;
#pragma unroll
  for (int rr = ty; rr < 32; rr += 8)
    tile[rr][tx] = src[(long)(r0 + rr) * C + c0 + tx];
  __syncthreads();
#pragma unroll
  for (int cc = ty; cc < 32; cc += 8)
    dst[(long)(c0 + cc) * R + r0 + tx] = tobf(tile[tx][cc]);
}

// ---------------------------------------------------------------------------
// Batched GEMM on pre-packed bf16: C[b] = alpha * (A[b] (MxK) @ Bt[b]^T),
// Bt stored N x K (transposed). One wave computes 16 x (NT*16).
// grid = (N/(16*NT), M/16, batch), blockDim = 32.
// cmode: 0 = f32 row-major, 1 = bf16 row-major, 2 = bf16 transposed (NxM).
template <int NT>
__global__ void __launch_bounds__(32, 1)
wmma_gemm_bf(const __bf16* __restrict__ A,
             const __bf16* __restrict__ Bt,
             void* __restrict__ Cout,
             int K, int lda, int ldbt, int ldc,
             long batA, long batB, long batC,
             float alpha, int cmode)
{
  A  += (long)blockIdx.z * batA;
  Bt += (long)blockIdx.z * batB;

  const int lane = threadIdx.x & 31;
  const int n    = lane & 15;
  const int half = lane >> 4;
  const int khA  = half << 4;    // A: upper half holds K 16..31
  const int khB  = half << 4;    // B: upper half holds K 16..31

  const long row0 = (long)blockIdx.y << 4;
  const long col0 = (long)blockIdx.x * (NT * 16);

  v8f acc[NT];
  const v8f vzero = {0.f, 0.f, 0.f, 0.f, 0.f, 0.f, 0.f, 0.f};
#pragma unroll
  for (int c = 0; c < NT; ++c) acc[c] = vzero;

  const __bf16* arow = A + (row0 + n) * (long)lda + khA;
  const __bf16* brow[NT];
#pragma unroll
  for (int c = 0; c < NT; ++c)
    brow[c] = Bt + (col0 + c * 16 + n) * (long)ldbt + khB;

  for (int k0 = 0; k0 < K; k0 += 32) {
    __builtin_prefetch(arow + k0 + 128, 0, 0);
    const v16bf af = ld_frag(arow + k0, arow + k0 + 8);
#pragma unroll
    for (int c = 0; c < NT; ++c) {
      const v16bf bf = ld_frag(brow[c] + k0, brow[c] + k0 + 8);
      acc[c] = __builtin_amdgcn_wmma_f32_16x16x32_bf16(false, af, false, bf,
                                                       (short)0, acc[c], false, false);
    }
  }

  if (cmode == 0) {
    float* C = (float*)Cout + (long)blockIdx.z * batC;
#pragma unroll
    for (int c = 0; c < NT; ++c)
#pragma unroll
      for (int i = 0; i < 8; ++i)
        C[(row0 + i + 8 * half) * (long)ldc + col0 + c * 16 + n] = alpha * acc[c][i];
  } else if (cmode == 1) {
    __bf16* C = (__bf16*)Cout + (long)blockIdx.z * batC;
#pragma unroll
    for (int c = 0; c < NT; ++c)
#pragma unroll
      for (int i = 0; i < 8; ++i)
        C[(row0 + i + 8 * half) * (long)ldc + col0 + c * 16 + n] = tobf(alpha * acc[c][i]);
  } else {
    __bf16* C = (__bf16*)Cout + (long)blockIdx.z * batC;
#pragma unroll
    for (int c = 0; c < NT; ++c)
#pragma unroll
      for (int i = 0; i < 8; ++i)
        C[(col0 + c * 16 + n) * (long)ldc + row0 + i + 8 * half] = tobf(alpha * acc[c][i]);
  }
}

// ---------------------------------------------------------------------------
// In-place RMS norm over first `ncols` of each row (f32, row stride `ld`).
__global__ void __launch_bounds__(256)
rms_inplace(float* __restrict__ x, const float* __restrict__ g, int ncols, int ld)
{
  float* xr = x + (long)blockIdx.x * ld;
  __shared__ float red[8];
  __shared__ float rsh;

  float s = 0.f;
  for (int c = threadIdx.x; c < ncols; c += 256) {
    float v = xr[c];
    s += v * v;
  }
#pragma unroll
  for (int off = 16; off > 0; off >>= 1) s += __shfl_xor(s, off, 32);
  if ((threadIdx.x & 31) == 0) red[threadIdx.x >> 5] = s;
  __syncthreads();
  if (threadIdx.x == 0) {
    float t = 0.f;
#pragma unroll
    for (int i = 0; i < 8; ++i) t += red[i];
    rsh = rsqrtf(t / (float)ncols + 1e-5f);
  }
  __syncthreads();
  const float r = rsh;
  for (int c = threadIdx.x; c < ncols; c += 256) xr[c] = xr[c] * r * g[c];
}

// ---------------------------------------------------------------------------
// In-place RoPE on k_pe (cols 256..287 of f32 latent). grid = T, block = 16.
__global__ void __launch_bounds__(32)
rope_k_kernel(float* __restrict__ latent, const int* __restrict__ pos)
{
  const int t = blockIdx.x, j = threadIdx.x;
  const float p = (float)pos[t];
  const float inv = __powf(10000.f, -(float)j / 16.f);
  float sn, cs;
  __sincosf(p * inv, &sn, &cs);
  float* kp = latent + (long)t * DKV_ + KVLORA_;
  const float x1 = kp[j], x2 = kp[j + 16];
  kp[j]      = x1 * cs - x2 * sn;
  kp[j + 16] = x2 * cs + x1 * sn;
}

// In-place RoPE (+score scale) on q_pe in f32 qbuf. grid=(T,H), block=16.
__global__ void __launch_bounds__(32)
rope_q_kernel(float* __restrict__ qbuf, const int* __restrict__ pos, float scale)
{
  const int t = blockIdx.x, h = blockIdx.y, j = threadIdx.x;
  const float p = (float)pos[t];
  const float inv = __powf(10000.f, -(float)j / 16.f);
  float sn, cs;
  __sincosf(p * inv, &sn, &cs);
  float* qp = qbuf + (long)t * (H_ * DQ_) + h * DQ_ + NOPE_;
  const float x1 = qp[j], x2 = qp[j + 16];
  qp[j]      = (x1 * cs - x2 * sn) * scale;
  qp[j + 16] = (x2 * cs + x1 * sn) * scale;
}

// ---------------------------------------------------------------------------
// Flash attention (MLA absorbed), all operands pre-packed bf16.
// Q tile (16 x 288) is staged in LDS once and score WMMAs read their
// A-fragments via ds_load_b128 -> register pressure ~110 VGPRs, no spills.
//   qlat : T x (H*256)  bf16, score-scaled
//   qpe  : T x (H*96)   bf16, rope part at h*96+64, score-scaled
//   kin  : T x 288      bf16 (rms v_in ++ roped k_pe)
//   vhT  : (H*64) x T   bf16 (v_in @ w_vc[h], transposed)
//   obuf : T x (H*64)   bf16 out
// grid = (T/16, H), blockDim = 32 (one wave).
__global__ void __launch_bounds__(32, 1)
mla_flash(const __bf16* __restrict__ qlat,
          const __bf16* __restrict__ qpe,
          const __bf16* __restrict__ kin,
          const __bf16* __restrict__ vhT,
          __bf16* __restrict__ obuf)
{
  const int qt   = blockIdx.x;
  const int h    = blockIdx.y;
  const int lane = threadIdx.x & 31;
  const int n    = lane & 15;
  const int half = lane >> 4;
  const int khA  = half << 4;
  const int khB  = half << 4;
  const int t0   = qt << 4;

  __shared__ alignas(16) __bf16 qls[16][DKV_];  // 16 x 288 staged Q tile
  __shared__ alignas(16) __bf16 pls[16][32];    // P tile C-layout -> A-layout

  // --- cooperative stage of Q tile into LDS (b128 in, b128 out) ---
  {
    const __bf16* qbase = qlat + (long)t0 * (H_ * KVLORA_) + h * KVLORA_;
    for (int idx = lane; idx < 512; idx += 32) {   // 16 rows x 32 uint4
      const int r = idx >> 5, c = idx & 31;
      reinterpret_cast<uint4*>(&qls[r][0])[c] =
          *reinterpret_cast<const uint4*>(qbase + (long)r * (H_ * KVLORA_) + c * 8);
    }
    const __bf16* pbase = qpe + (long)t0 * (H_ * DQ_) + h * DQ_ + NOPE_;
    for (int idx = lane; idx < 64; idx += 32) {    // 16 rows x 4 uint4
      const int r = idx >> 2, c = idx & 3;
      reinterpret_cast<uint4*>(&qls[r][KVLORA_])[c] =
          *reinterpret_cast<const uint4*>(pbase + (long)r * (H_ * DQ_) + c * 8);
    }
  }
  __syncthreads();

  float mrow[8], lrow[8];
  v8f oacc[4];
  const v8f vzero = {0.f, 0.f, 0.f, 0.f, 0.f, 0.f, 0.f, 0.f};
#pragma unroll
  for (int i = 0; i < 8; ++i) { mrow[i] = -1e30f; lrow[i] = 0.f; }
#pragma unroll
  for (int c = 0; c < 4; ++c) oacc[c] = vzero;

  const int jmax = (t0 + 15) >> 5;
  for (int j = 0; j <= jmax; ++j) {
    const int s0 = j << 5;

    // --- S = Q . K^T for 32 keys -> two 16x16 C fragments ---
    v8f sc0 = vzero, sc1 = vzero;
    const __bf16* k0p = kin + (long)(s0 + n) * DKV_ + khB;
    const __bf16* k1p = kin + (long)(s0 + 16 + n) * DKV_ + khB;
#pragma unroll
    for (int d = 0; d < 9; ++d) {
      const v16bf af = ld_frag(&qls[n][d * 32 + khA], &qls[n][d * 32 + khA + 8]);
      const v16bf b0 = ld_frag(k0p + d * 32, k0p + d * 32 + 8);
      sc0 = __builtin_amdgcn_wmma_f32_16x16x32_bf16(false, af, false, b0,
                                                    (short)0, sc0, false, false);
      const v16bf b1 = ld_frag(k1p + d * 32, k1p + d * 32 + 8);
      sc1 = __builtin_amdgcn_wmma_f32_16x16x32_bf16(false, af, false, b1,
                                                    (short)0, sc1, false, false);
    }

    // --- causal mask + online softmax ---
#pragma unroll
    for (int i = 0; i < 8; ++i) {
      const int row = t0 + i + 8 * half;
      float s0v = (s0 + n      <= row) ? sc0[i] : -1e30f;
      float s1v = (s0 + 16 + n <= row) ? sc1[i] : -1e30f;
      float mx = fmaxf(s0v, s1v);
#pragma unroll
      for (int off = 1; off < 16; off <<= 1)
        mx = fmaxf(mx, __shfl_xor(mx, off, 16));
      const float mn = fmaxf(mrow[i], mx);
      const float al = __expf(mrow[i] - mn);
      mrow[i] = mn;
      const float p0 = __expf(s0v - mn);
      const float p1 = __expf(s1v - mn);
      float ps = p0 + p1;
#pragma unroll
      for (int off = 1; off < 16; off <<= 1)
        ps += __shfl_xor(ps, off, 16);
      lrow[i] = lrow[i] * al + ps;
      pls[i + 8 * half][n]      = tobf(p0);
      pls[i + 8 * half][16 + n] = tobf(p1);
#pragma unroll
      for (int c = 0; c < 4; ++c) oacc[c][i] = oacc[c][i] * al;
    }
    __syncthreads();

    // --- P (16x32) as A-fragment straight from LDS (b128 reads) ---
    const v16bf pf = ld_frag(&pls[n][khA], &pls[n][khA + 8]);

    // --- O += P @ Vh (4 column tiles, V pre-transposed -> contiguous) ---
#pragma unroll
    for (int ct = 0; ct < 4; ++ct) {
      const __bf16* vrow = vhT + (long)(h * V_ + ct * 16 + n) * T_ + s0 + khB;
      const v16bf bf = ld_frag(vrow, vrow + 8);
      oacc[ct] = __builtin_amdgcn_wmma_f32_16x16x32_bf16(false, pf, false, bf,
                                                         (short)0, oacc[ct], false, false);
    }
    __syncthreads();
  }

  // epilogue: normalize and store bf16
#pragma unroll
  for (int ct = 0; ct < 4; ++ct)
#pragma unroll
    for (int i = 0; i < 8; ++i)
      obuf[(long)(t0 + i + 8 * half) * (H_ * V_) + h * V_ + ct * 16 + n] =
          tobf(oacc[ct][i] / lrow[i]);
}

// ---------------------------------------------------------------------------
extern "C" void kernel_launch(void* const* d_in, const int* in_sizes, int n_in,
                              void* d_out, int out_size, void* d_ws, size_t ws_size,
                              hipStream_t stream) {
  const int*   pos    = (const int*)  d_in[0];
  const float* hidden = (const float*)d_in[1];   // T x 2560
  const float* w_qa   = (const float*)d_in[2];   // 2560 x 768
  const float* g_qa   = (const float*)d_in[3];
  const float* w_qb   = (const float*)d_in[4];   // 768 x 3840
  const float* w_kva  = (const float*)d_in[5];   // 2560 x 288
  const float* g_kva  = (const float*)d_in[6];
  const float* w_kc   = (const float*)d_in[7];   // 40 x 64 x 256
  const float* w_vc   = (const float*)d_in[8];   // 40 x 256 x 64
  const float* w_o    = (const float*)d_in[9];   // 2560 x 2560
  float* out = (float*)d_out;

  char* p = (char*)d_ws;
  float*  qa_f   = (float*)p;  p += (size_t)T_ * QLORA_ * 4;
  float*  qbuf_f = (float*)p;  p += (size_t)T_ * H_ * DQ_ * 4;
  float*  lat_f  = (float*)p;  p += (size_t)T_ * DKV_ * 4;
  __bf16* hid_b  = (__bf16*)p; p += (size_t)T_ * HID_ * 2;
  __bf16* qa_b   = (__bf16*)p; p += (size_t)T_ * QLORA_ * 2;
  __bf16* qbuf_b = (__bf16*)p; p += (size_t)T_ * H_ * DQ_ * 2;
  __bf16* kin_b  = (__bf16*)p; p += (size_t)T_ * DKV_ * 2;
  __bf16* qlat_b = (__bf16*)p; p += (size_t)T_ * H_ * KVLORA_ * 2;
  __bf16* vhT_b  = (__bf16*)p; p += (size_t)H_ * V_ * T_ * 2;
  __bf16* obuf_b = (__bf16*)p; p += (size_t)T_ * H_ * V_ * 2;
  __bf16* wqa_t  = (__bf16*)p; p += (size_t)HID_ * QLORA_ * 2;
  __bf16* wqb_t  = (__bf16*)p; p += (size_t)QLORA_ * H_ * DQ_ * 2;
  __bf16* wkva_t = (__bf16*)p; p += (size_t)HID_ * DKV_ * 2;
  __bf16* wkc_t  = (__bf16*)p; p += (size_t)H_ * NOPE_ * KVLORA_ * 2;
  __bf16* wvc_t  = (__bf16*)p; p += (size_t)H_ * KVLORA_ * V_ * 2;
  __bf16* wo_t   = (__bf16*)p; p += (size_t)HID_ * HID_ * 2;

  const dim3 wv(32);
  const dim3 b256(256);

  // --- weight / activation packing (bf16; weights transposed to N x K) ---
  cvt_rm<<<dim3((T_ * HID_ / 4 + 255) / 256), b256, 0, stream>>>(
      hidden, hid_b, (long)T_ * HID_ / 4);
  cvt_tr<<<dim3(QLORA_ / 32, HID_ / 32, 1), b256, 0, stream>>>(
      w_qa, wqa_t, HID_, QLORA_, 0, 0);
  cvt_tr<<<dim3(H_ * DQ_ / 32, QLORA_ / 32, 1), b256, 0, stream>>>(
      w_qb, wqb_t, QLORA_, H_ * DQ_, 0, 0);
  cvt_tr<<<dim3(DKV_ / 32, HID_ / 32, 1), b256, 0, stream>>>(
      w_kva, wkva_t, HID_, DKV_, 0, 0);
  cvt_tr<<<dim3(KVLORA_ / 32, NOPE_ / 32, H_), b256, 0, stream>>>(
      w_kc, wkc_t, NOPE_, KVLORA_, (long)NOPE_ * KVLORA_, (long)NOPE_ * KVLORA_);
  cvt_tr<<<dim3(V_ / 32, KVLORA_ / 32, H_), b256, 0, stream>>>(
      w_vc, wvc_t, KVLORA_, V_, (long)KVLORA_ * V_, (long)KVLORA_ * V_);
  cvt_tr<<<dim3(HID_ / 32, HID_ / 32, 1), b256, 0, stream>>>(
      w_o, wo_t, HID_, HID_, 0, 0);

  // --- 1) qa = hidden @ w_qa ; rms ; pack ---
  wmma_gemm_bf<4><<<dim3(QLORA_ / 64, T_ / 16, 1), wv, 0, stream>>>(
      hid_b, wqa_t, qa_f, HID_, HID_, HID_, QLORA_, 0, 0, 0, 1.f, 0);
  rms_inplace<<<dim3(T_), b256, 0, stream>>>(qa_f, g_qa, QLORA_, QLORA_);
  cvt_rm<<<dim3((T_ * QLORA_ / 4 + 255) / 256), b256, 0, stream>>>(
      qa_f, qa_b, (long)T_ * QLORA_ / 4);

  // --- 2) q = qa @ w_qb ; rope+scale q_pe ; pack ---
  wmma_gemm_bf<4><<<dim3(H_ * DQ_ / 64, T_ / 16, 1), wv, 0, stream>>>(
      qa_b, wqb_t, qbuf_f, QLORA_, QLORA_, QLORA_, H_ * DQ_, 0, 0, 0, 1.f, 0);
  rope_q_kernel<<<dim3(T_, H_), dim3(16), 0, stream>>>(qbuf_f, pos, SCALE_);
  cvt_rm<<<dim3((T_ * H_ * DQ_ / 4 + 255) / 256), b256, 0, stream>>>(
      qbuf_f, qbuf_b, (long)T_ * H_ * DQ_ / 4);

  // --- 3) latent = hidden @ w_kva ; rms(256) ; rope k_pe ; pack -> k_in ---
  wmma_gemm_bf<2><<<dim3(DKV_ / 32, T_ / 16, 1), wv, 0, stream>>>(
      hid_b, wkva_t, lat_f, HID_, HID_, HID_, DKV_, 0, 0, 0, 1.f, 0);
  rms_inplace<<<dim3(T_), b256, 0, stream>>>(lat_f, g_kva, KVLORA_, DKV_);
  rope_k_kernel<<<dim3(T_), dim3(16), 0, stream>>>(lat_f, pos);
  cvt_rm<<<dim3((T_ * DKV_ / 4 + 255) / 256), b256, 0, stream>>>(
      lat_f, kin_b, (long)T_ * DKV_ / 4);

  // --- 4) q_lat[h] = scale * (q_nope[:,h,:] @ w_kc[h])  -> bf16 ---
  wmma_gemm_bf<4><<<dim3(KVLORA_ / 64, T_ / 16, H_), wv, 0, stream>>>(
      qbuf_b, wkc_t, qlat_b, NOPE_, H_ * DQ_, NOPE_, H_ * KVLORA_,
      /*batA=*/DQ_, /*batB=*/(long)NOPE_ * KVLORA_, /*batC=*/KVLORA_, SCALE_, 1);

  // --- 5) vhT[h] = (v_in @ w_vc[h])^T  -> bf16 (H*64) x T ---
  wmma_gemm_bf<4><<<dim3(V_ / 64, T_ / 16, H_), wv, 0, stream>>>(
      kin_b, wvc_t, vhT_b, KVLORA_, DKV_, KVLORA_, T_,
      /*batA=*/0, /*batB=*/(long)KVLORA_ * V_, /*batC=*/(long)V_ * T_, 1.f, 2);

  // --- 6) flash attention -> obuf (bf16 T x 2560) ---
  mla_flash<<<dim3(T_ / 16, H_), wv, 0, stream>>>(qlat_b, qbuf_b, kin_b, vhT_b, obuf_b);

  // --- 7) out = obuf @ w_o (f32) ---
  wmma_gemm_bf<4><<<dim3(HID_ / 64, T_ / 16, 1), wv, 0, stream>>>(
      obuf_b, wo_t, out, H_ * V_, H_ * V_, H_ * V_, HID_, 0, 0, 0, 1.f, 0);
}
